// GatedCrossModalAttention_81192061764396
// MI455X (gfx1250) — matrix operand order repaired and checked by reading, not compiled, not run on hardware
//
#include <hip/hip_runtime.h>
#include <hip/hip_bf16.h>

typedef __attribute__((ext_vector_type(16))) __bf16 v16bf;
typedef __attribute__((ext_vector_type(8)))  float  v8f;

union ABFrag { v16bf v; uint4 q[2]; };

__device__ __forceinline__ unsigned short f2bf(float f) {
  union { float f; unsigned u; } c; c.f = f;
  unsigned u = c.u;
  u += 0x7FFFu + ((u >> 16) & 1u);          // round-to-nearest-even
  return (unsigned short)(u >> 16);
}
__device__ __forceinline__ float bf2f(unsigned short h) {
  union { unsigned u; float f; } c; c.u = ((unsigned)h) << 16; return c.f;
}

// Two 16x16 output tiles (rows 0-15 and 16-31 of a 32-row panel) sharing each
// B fragment: 1 B load feeds 2 WMMAs -> halves L2 weight traffic vs 1 row-tile.
// aRow0/aRow1: LDS pointers to this lane's A rows (row = lane&15, 16+lane&15).
// bRow: global bf16 weight [in][out] at (krow=(lane&15)+16*kg)*bOut + nBase.
__device__ __forceinline__ void gemm_tile2(v8f& acc0, v8f& acc1,
                                           const unsigned short* aRow0,
                                           const unsigned short* aRow1,
                                           const unsigned short* bRow,
                                           int bOut, int kLen, int kg) {
  const unsigned short* a0 = aRow0 + 8 * kg;
  const unsigned short* a1 = aRow1 + 8 * kg;
  for (int k = 0; k < kLen; k += 32) {
    ABFrag fa0, fa1, fb;
    fa0.q[0] = *(const uint4*)(a0 + k);      // K = k + 8*kg + 0..7
    fa0.q[1] = *(const uint4*)(a0 + k + 16); // K = k + 16 + 8*kg + 0..7
    fa1.q[0] = *(const uint4*)(a1 + k);
    fa1.q[1] = *(const uint4*)(a1 + k + 16);
    {
      const unsigned short* bc = bRow - (size_t)((threadIdx.x & 15) + 16 * kg) * bOut + (threadIdx.x & 15);
      unsigned short* fbs = (unsigned short*)&fb;
      for (int s = 0; s < 8; ++s) {
        fbs[s]     = bc[(size_t)(8 * kg + s) * bOut];
        fbs[8 + s] = bc[(size_t)(16 + 8 * kg + s) * bOut];
      }
    }
    bRow += (size_t)32 * bOut;
    __builtin_prefetch(bRow, 0, 1);          // global_prefetch_b8 next K-panel
    acc0 = __builtin_amdgcn_wmma_f32_16x16x32_bf16(false, fa0.v, false, fb.v,
                                                   (short)0, acc0, false, false);
    acc1 = __builtin_amdgcn_wmma_f32_16x16x32_bf16(false, fa1.v, false, fb.v,
                                                   (short)0, acc1, false, false);
  }
}

// Weight prep: WT[m][i][o] = bf16(W[m][o][i])  (torch Linear W is [out][in])
__global__ __launch_bounds__(256)
void wtrans(const float* __restrict__ W, unsigned short* __restrict__ WT,
            int Mn, int OUT, int IN) {
  int total = Mn * OUT * IN;
  for (int idx = blockIdx.x * 256 + threadIdx.x; idx < total;
       idx += gridDim.x * 256) {
    int m = idx / (OUT * IN);
    int r = idx - m * OUT * IN;
    int i = r / OUT;
    int o = r - i * OUT;
    WT[idx] = f2bf(W[(size_t)m * OUT * IN + (size_t)o * IN + i]);
  }
}

#define Bdim 32768
#define Ddim 256
#define TT   32          // tokens per block
#define NT   512         // threads per block (16 wave32)

__global__ __launch_bounds__(NT)
void gcma_fused(const float* __restrict__ x,
                const float* __restrict__ bq, const float* __restrict__ bk,
                const float* __restrict__ bv, const float* __restrict__ bo,
                const float* __restrict__ gb1, const float* __restrict__ gW2,
                const float* __restrict__ gb2,
                const float* __restrict__ gamma, const float* __restrict__ beta,
                const float* __restrict__ fb1, const float* __restrict__ fb2,
                const unsigned short* __restrict__ WqT, const unsigned short* __restrict__ WkT,
                const unsigned short* __restrict__ WvT, const unsigned short* __restrict__ WoT,
                const unsigned short* __restrict__ gW1T, const unsigned short* __restrict__ fW1T,
                const unsigned short* __restrict__ fW2T,
                float* __restrict__ outU, float* __restrict__ outG) {
  // Padded row strides (264/520/1032 elems) shift banks by 4 per row ->
  // conflict-free 16-row column-group access for WMMA fragment loads/stores.
  __shared__ __align__(16) unsigned short lxbf[4][TT][264];      // ctx bf16 (A)
  __shared__ __align__(16) unsigned short Qt[TT][264];           // Q, then o
  __shared__ __align__(16) unsigned short kvpool[2][4][TT][264]; // K,V; reused as hff
  __shared__ __align__(16) unsigned short cat[TT][520];          // [x || attn]
  __shared__ __align__(16) unsigned short nrm[TT][264];          // normed bf16
  __shared__ __align__(16) float yf[TT][260];                    // y->normed->updated
  __shared__ float ghs[TT][64];
  __shared__ float gateS[TT], muS[TT], rsS[TT];

  auto& Kt = kvpool[0];
  auto& Vt = kvpool[1];
  unsigned short* hffp = &kvpool[0][0][0][0];                    // [TT][1032] alias
#define HFF(r, c) hffp[(r) * 1032 + (c)]

  const int t0   = blockIdx.x * TT;
  const int tid  = threadIdx.x;
  const int lane = tid & 31;
  const int wv   = tid >> 5;      // 16 waves
  const int lrow = lane & 15;
  const int kg   = lane >> 4;

  // Stage shared context: x[c, tile32, :] for c=0..3, fp32 -> bf16 (coalesced)
  for (int idx = tid; idx < 4 * TT * Ddim; idx += NT) {
    int c = idx >> 13, r = (idx >> 8) & 31, d = idx & 255;
    lxbf[c][r][d] = f2bf(x[((size_t)c * Bdim + t0 + r) * Ddim + d]);
  }
  __syncthreads();

  for (int m = 0; m < 4; ++m) {
    // cat[:, :256] = x[m]  (first half of gate input)
    for (int idx = tid; idx < TT * Ddim; idx += NT) {
      int r = idx >> 8, d = idx & 255;
      cat[r][d] = lxbf[m][r][d];
    }

    // ---- Q = x[m] @ WqT + bq  (16 col-tiles, 1 per wave) ---------------
    {
      int nb = wv * 16;
      v8f a0 = {}, a1 = {};
      gemm_tile2(a0, a1, &lxbf[m][lrow][0], &lxbf[m][16 + lrow][0],
                 WqT + (size_t)m * 65536 + (size_t)(lrow + 16 * kg) * 256 + nb,
                 256, 256, kg);
      float bias = bq[m * Ddim + nb + lrow];
      for (int v = 0; v < 8; ++v) {
        Qt[v + 8 * kg][nb + lrow]      = f2bf(a0[v] + bias);
        Qt[16 + v + 8 * kg][nb + lrow] = f2bf(a1[v] + bias);
      }
    }

    // ---- K_c, V_c = x[c] @ W{k,v}T + b   (128 col-tile jobs) -----------
    for (int job = wv; job < 128; job += 16) {
      int ct = job & 15, c = (job >> 4) & 3, kv = job >> 6;
      int nb = ct * 16;
      const unsigned short* Wt = (kv ? WvT : WkT) + (size_t)m * 65536;
      const float* bb = kv ? bv : bk;
      v8f a0 = {}, a1 = {};
      gemm_tile2(a0, a1, &lxbf[c][lrow][0], &lxbf[c][16 + lrow][0],
                 Wt + (size_t)(lrow + 16 * kg) * 256 + nb, 256, 256, kg);
      float bias = bb[m * Ddim + nb + lrow];
      unsigned short (*dst)[TT][264] = kv ? Vt : Kt;
      for (int v = 0; v < 8; ++v) {
        dst[c][v + 8 * kg][nb + lrow]      = f2bf(a0[v] + bias);
        dst[c][16 + v + 8 * kg][nb + lrow] = f2bf(a1[v] + bias);
      }
    }
    __syncthreads();

    // ---- scores / softmax over C=4 / weighted V  (VALU; tiny) ----------
    if (tid < TT * 8) {
      int b = tid >> 3, h = tid & 7, off = h * 32;
      float q[32];
      for (int j = 0; j < 32; ++j) q[j] = bf2f(Qt[b][off + j]);
      float s[4], mx = -1e30f;
      for (int c = 0; c < 4; ++c) {
        float d0 = 0.f;
        for (int j = 0; j < 32; ++j) d0 += q[j] * bf2f(Kt[c][b][off + j]);
        s[c] = d0 * 0.17677669529663687f;   // 1/sqrt(dh=32)
        mx = fmaxf(mx, s[c]);
      }
      float den = 0.f;
      for (int c = 0; c < 4; ++c) { s[c] = __expf(s[c] - mx); den += s[c]; }
      float inv = 1.0f / den;
      for (int j = 0; j < 32; ++j) {
        float o = 0.f;
        for (int c = 0; c < 4; ++c) o += s[c] * bf2f(Vt[c][b][off + j]);
        Qt[b][off + j] = f2bf(o * inv);     // reuse Qt as o-buffer
      }
    }
    __syncthreads();

    // ---- attn = o @ WoT + bo  ->  cat[:, 256:512] ----------------------
    {
      int nb = wv * 16;
      v8f a0 = {}, a1 = {};
      gemm_tile2(a0, a1, &Qt[lrow][0], &Qt[16 + lrow][0],
                 WoT + (size_t)m * 65536 + (size_t)(lrow + 16 * kg) * 256 + nb,
                 256, 256, kg);
      float bias = bo[m * Ddim + nb + lrow];
      for (int v = 0; v < 8; ++v) {
        cat[v + 8 * kg][256 + nb + lrow]      = f2bf(a0[v] + bias);
        cat[16 + v + 8 * kg][256 + nb + lrow] = f2bf(a1[v] + bias);
      }
    }
    __syncthreads();

    // ---- gate hidden: tanh([x||attn] @ gW1T + gb1), 4 col tiles --------
    if (wv < 4) {
      int nb = wv * 16;
      v8f a0 = {}, a1 = {};
      gemm_tile2(a0, a1, &cat[lrow][0], &cat[16 + lrow][0],
                 gW1T + (size_t)m * (512 * 64) + (size_t)(lrow + 16 * kg) * 64 + nb,
                 64, 512, kg);
      float bias = gb1[m * 64 + nb + lrow];
      for (int v = 0; v < 8; ++v) {
        ghs[v + 8 * kg][nb + lrow]      = tanhf(a0[v] + bias);
        ghs[16 + v + 8 * kg][nb + lrow] = tanhf(a1[v] + bias);
      }
    }
    __syncthreads();
    if (tid < TT) {
      float g = gb2[m];
      for (int j = 0; j < 64; ++j) g += ghs[tid][j] * gW2[m * 64 + j];
      g = 1.0f / (1.0f + __expf(-g));
      gateS[tid] = g;
      outG[(size_t)m * Bdim + t0 + tid] = g;
    }
    __syncthreads();

    // ---- y = x + gate*attn ---------------------------------------------
    for (int idx = tid; idx < TT * Ddim; idx += NT) {
      int r = idx >> 8, d = idx & 255;
      float att = bf2f(cat[r][256 + d]);
      float xv = x[((size_t)m * Bdim + t0 + r) * Ddim + d];
      yf[r][d] = xv + gateS[r] * att;
    }
    __syncthreads();

    // ---- LayerNorm ------------------------------------------------------
    if (tid < TT) {
      float mu = 0.f;
      for (int d = 0; d < 256; ++d) mu += yf[tid][d];
      mu *= (1.0f / 256.0f);
      float var = 0.f;
      for (int d = 0; d < 256; ++d) { float t = yf[tid][d] - mu; var += t * t; }
      muS[tid] = mu;
      rsS[tid] = rsqrtf(var * (1.0f / 256.0f) + 1e-5f);
    }
    __syncthreads();
    for (int idx = tid; idx < TT * Ddim; idx += NT) {
      int r = idx >> 8, d = idx & 255;
      float n = (yf[r][d] - muS[r]) * rsS[r] * gamma[m * Ddim + d] + beta[m * Ddim + d];
      nrm[r][d] = f2bf(n);
      yf[r][d] = n;                         // keep f32 normed for final residual
    }
    __syncthreads();

    // ---- FF1: gelu(normed @ fW1T + fb1) -> hff[32][1024] ---------------
    for (int ct = wv; ct < 64; ct += 16) {
      int nb = ct * 16;
      v8f a0 = {}, a1 = {};
      gemm_tile2(a0, a1, &nrm[lrow][0], &nrm[16 + lrow][0],
                 fW1T + (size_t)m * (256 * 1024) + (size_t)(lrow + 16 * kg) * 1024 + nb,
                 1024, 256, kg);
      for (int rt = 0; rt < 2; ++rt) {
        v8f& acc = rt ? a1 : a0;
        for (int v = 0; v < 8; ++v) {
          float h0 = acc[v] + fb1[m * 1024 + nb + lrow];
          float t = h0 + 0.044715f * h0 * h0 * h0;      // tanh-approx GELU
          float gl = 0.5f * h0 * (1.0f + tanhf(0.7978845608028654f * t));
          HFF(16 * rt + v + 8 * kg, nb + lrow) = f2bf(gl);
        }
      }
    }
    __syncthreads();

    // ---- FF2 + residual: updated = normed + hff @ fW2T + fb2 -----------
    {
      int nb = wv * 16;
      v8f a0 = {}, a1 = {};
      gemm_tile2(a0, a1, &HFF(lrow, 0), &HFF(16 + lrow, 0),
                 fW2T + (size_t)m * (1024 * 256) + (size_t)(lrow + 16 * kg) * 256 + nb,
                 256, 1024, kg);
      float bias = fb2[m * Ddim + nb + lrow];
      for (int v = 0; v < 8; ++v) {
        yf[v + 8 * kg][nb + lrow]      += a0[v] + bias;
        yf[16 + v + 8 * kg][nb + lrow] += a1[v] + bias;
      }
    }
    __syncthreads();
    for (int idx = tid; idx < TT * Ddim; idx += NT) {   // coalesced store
      int r = idx >> 8, d = idx & 255;
      outU[((size_t)m * Bdim + t0 + r) * Ddim + d] = yf[r][d];
    }
    __syncthreads();
  }
#undef HFF
}

extern "C" void kernel_launch(void* const* d_in, const int* in_sizes, int n_in,
                              void* d_out, int out_size, void* d_ws, size_t ws_size,
                              hipStream_t stream) {
  (void)in_sizes; (void)n_in; (void)out_size; (void)ws_size;
  const float* x    = (const float*)d_in[0];
  const float* Wq   = (const float*)d_in[1];
  const float* bq   = (const float*)d_in[2];
  const float* Wk   = (const float*)d_in[3];
  const float* bk   = (const float*)d_in[4];
  const float* Wv   = (const float*)d_in[5];
  const float* bv   = (const float*)d_in[6];
  const float* Wo   = (const float*)d_in[7];
  const float* bo   = (const float*)d_in[8];
  const float* gW1  = (const float*)d_in[9];
  const float* gb1  = (const float*)d_in[10];
  const float* gW2  = (const float*)d_in[11];
  const float* gb2  = (const float*)d_in[12];
  const float* gamma= (const float*)d_in[13];
  const float* beta = (const float*)d_in[14];
  const float* fW1  = (const float*)d_in[15];
  const float* fb1  = (const float*)d_in[16];
  const float* fW2  = (const float*)d_in[17];
  const float* fb2  = (const float*)d_in[18];

  unsigned short* ws  = (unsigned short*)d_ws;     // ~6.6 MB bf16 weights
  unsigned short* WqT = ws;
  unsigned short* WkT = WqT + 262144;              // 4*256*256
  unsigned short* WvT = WkT + 262144;
  unsigned short* WoT = WvT + 262144;
  unsigned short* gW1T= WoT + 262144;              // 4*512*64
  unsigned short* fW1T= gW1T + 131072;             // 4*256*1024
  unsigned short* fW2T= fW1T + 1048576;            // 4*1024*256

  wtrans<<<512, 256, 0, stream>>>(Wq, WqT, 4, 256, 256);
  wtrans<<<512, 256, 0, stream>>>(Wk, WkT, 4, 256, 256);
  wtrans<<<512, 256, 0, stream>>>(Wv, WvT, 4, 256, 256);
  wtrans<<<512, 256, 0, stream>>>(Wo, WoT, 4, 256, 256);
  wtrans<<<256, 256, 0, stream>>>(gW1, gW1T, 4, 64, 512);
  wtrans<<<1024, 256, 0, stream>>>(fW1, fW1T, 4, 1024, 256);
  wtrans<<<1024, 256, 0, stream>>>(fW2, fW2T, 4, 256, 1024);

  float* outU = (float*)d_out;
  float* outG = outU + (size_t)4 * Bdim * Ddim;
  gcma_fused<<<Bdim / TT, NT, 0, stream>>>(
      x, bq, bk, bv, bo, gb1, gW2, gb2, gamma, beta, fb1, fb2,
      WqT, WkT, WvT, WoT, gW1T, fW1T, fW2T, outU, outG);
}
